// GPT_17136919511183
// MI455X (gfx1250) — matrix operand order repaired and checked by reading, not compiled, not run on hardware
//
#include <hip/hip_runtime.h>
#include <hip/hip_bf16.h>
#include <math.h>

typedef __attribute__((ext_vector_type(16))) __bf16 v16bf;
typedef __attribute__((ext_vector_type(8)))  float  v8f;
typedef __attribute__((ext_vector_type(4)))  float  v4f;
typedef __attribute__((ext_vector_type(4)))  unsigned int v4u;
typedef __attribute__((ext_vector_type(8)))  int    v8i;
typedef __attribute__((ext_vector_type(4)))  int    v4i;
typedef unsigned short u16;

// Model constants (match reference)
constexpr int Cc = 1024, Hh = 16, Ll = 8, Bb = 2, Tt = 1024, Dd = 64;
constexpr int Mm = Bb * Tt;          // 2048 token rows
constexpr int Vv = 50257;            // vocab
constexpr int BH = Bb * Hh;          // 32 (batch*heads)

// Tensor Data Mover path: probe-confirmed builtins on ROCm 7.2 (5-arg) /
// amdgpu-toolchain (6-arg, detected via its TDM header).
#if defined(__gfx1250__) && defined(__HIP_DEVICE_COMPILE__) && \
    __has_builtin(__builtin_amdgcn_tensor_load_to_lds) &&      \
    __has_builtin(__builtin_amdgcn_s_wait_tensorcnt)
#define USE_TDM 1
#else
#define USE_TDM 0
#endif

__device__ __forceinline__ u16 f2bf(float f) {
  unsigned u = __float_as_uint(f);
  u += 0x7FFFu + ((u >> 16) & 1u);   // round-to-nearest-even
  return (u16)(u >> 16);
}

union FragAB { v16bf v; v4f f[2]; };

#if USE_TDM
// LDS byte offset of a __shared__ object (generic -> LDS addrspace cast).
__device__ __forceinline__ unsigned lds_off_of(void* p) {
  return (unsigned)(unsigned long long)(__attribute__((address_space(3))) void*)p;
}

// Issue one TDM 2-D tile load: `rows` x 32 bf16 elements, row stride
// `stride_elems`, zero-fill outside (rows, cols) per D# OOB rules.
// D# bit packing per CDNA5 ISA.
__device__ __forceinline__ void tdm_load_tile(unsigned lds_byte, const u16* gptr,
                                              unsigned rows, unsigned cols,
                                              unsigned stride_elems) {
  unsigned long long ga = (unsigned long long)gptr;
  v4u g0;
  g0.x = 1u;                                           // count=1, user descriptor
  g0.y = lds_byte;                                     // lds_addr
  g0.z = (unsigned)ga;                                 // global_addr[31:0]
  g0.w = (unsigned)((ga >> 32) & 0x01FFFFFFull)        // global_addr[56:32]
         | (2u << 30);                                 // type=2 ("image")
  v8i g1;
  g1[0] = (int)0x00010000u;                            // data_size=1 (2 bytes)
  g1[1] = (int)((cols & 0xFFFFu) << 16);               // tensor_dim0[15:0]
  g1[2] = (int)((cols >> 16) | ((rows & 0xFFFFu) << 16)); // dim0 hi | dim1 lo
  g1[3] = (int)((rows >> 16) | (32u << 16));           // dim1 hi | tile_dim0=32
  g1[4] = (int)128u;                                   // tile_dim1=128, tile_dim2=0
  g1[5] = (int)stride_elems;                           // tensor_dim0_stride lo32
  g1[6] = 0;                                           // stride hi | dim1_stride lo
  g1[7] = 0;
  v4i z4 = {0, 0, 0, 0};
#if __has_include(<hip/amd_detail/amd_gfx1250_TDM.h>)
  v8i z8 = {0, 0, 0, 0, 0, 0, 0, 0};
  __builtin_amdgcn_tensor_load_to_lds(g0, g1, z4, z4, z8, 0);
#else
  __builtin_amdgcn_tensor_load_to_lds(g0, g1, z4, z4, 0);
#endif
}
#endif  // USE_TDM

// ---------------------------------------------------------------------------
// f32 -> bf16 bulk convert (weights)
// ---------------------------------------------------------------------------
__global__ void cvt_k(const float* __restrict__ in, u16* __restrict__ out, size_t n) {
  for (size_t i = blockIdx.x * (size_t)blockDim.x + threadIdx.x; i < n;
       i += (size_t)gridDim.x * blockDim.x)
    out[i] = f2bf(in[i]);
}

// ---------------------------------------------------------------------------
// x = tok_emb[idx] + pos_emb  (f32 residual stream)
// ---------------------------------------------------------------------------
__global__ void embed_k(const int* __restrict__ idx, const float* __restrict__ tok,
                        const float* __restrict__ pos, float* __restrict__ X) {
  for (int i = blockIdx.x * blockDim.x + threadIdx.x; i < Mm * Cc;
       i += gridDim.x * blockDim.x) {
    int c = i & (Cc - 1);
    int bt = i / Cc;
    int t = bt & (Tt - 1);
    X[i] = tok[(size_t)idx[bt] * Cc + c] + pos[(size_t)t * Cc + c];
  }
}

// ---------------------------------------------------------------------------
// LayerNorm: f32 row -> bf16 row  (one block per row)
// ---------------------------------------------------------------------------
__global__ __launch_bounds__(256)
void ln_k(const float* __restrict__ X, const float* __restrict__ w,
          const float* __restrict__ b, u16* __restrict__ O) {
  const int row = blockIdx.x;
  const float* x = X + (size_t)row * Cc;
  u16* o = O + (size_t)row * Cc;
  __shared__ float r1[256], r2[256];
  const int tid = threadIdx.x;
  float s1 = 0.f, s2 = 0.f;
  for (int i = tid; i < Cc; i += 256) { float v = x[i]; s1 += v; s2 += v * v; }
  r1[tid] = s1; r2[tid] = s2; __syncthreads();
  for (int st = 128; st > 0; st >>= 1) {
    if (tid < st) { r1[tid] += r1[tid + st]; r2[tid] += r2[tid + st]; }
    __syncthreads();
  }
  const float mean = r1[0] * (1.0f / Cc);
  const float var  = r2[0] * (1.0f / Cc) - mean * mean;
  const float inv  = rsqrtf(var + 1e-5f);
  for (int i = tid; i < Cc; i += 256)
    o[i] = f2bf((x[i] - mean) * inv * w[i] + b[i]);
}

// ---------------------------------------------------------------------------
// WMMA bf16 GEMM:  Out[M,N] = A[M,K] @ Bw[N,K]^T + bias
// Block tile 128x128, 8 waves (4xM, 2xN), each wave 32x64 (2x4 WMMA frags).
// TDM path: wave 0 DMAs 128x32 A/B tiles into double-buffered LDS with
// tensor_load_to_lds, pipelined one K-slice ahead (s_wait_tensorcnt 2).
// Fallback: cooperative synchronous staging.
// EPI: 0=f32 store, 1=bf16 store, 2=f32 residual add, 3=erf-GELU -> bf16
// ---------------------------------------------------------------------------
constexpr int TILE_ELEMS = 128 * 32;
constexpr int TILE_BYTES = TILE_ELEMS * 2;

template<int EPI>
__global__ __launch_bounds__(256)
void gemm_bf16_k(const u16* __restrict__ A, const u16* __restrict__ Bw,
                 const float* __restrict__ bias, void* __restrict__ OutP,
                 int M, int N, int K, int ldo) {
  const int tid  = threadIdx.x;
  const int lane = tid & 31;
  const int wave = tid >> 5;
  const int bm = blockIdx.y * 128;
  const int bn = blockIdx.x * 128;
  const int wm = (wave & 3) * 32;
  const int wn = (wave >> 2) * 64;
  const int l15 = lane & 15;
  const int lh  = lane >> 4;

  const v8f vzero = {0.f, 0.f, 0.f, 0.f, 0.f, 0.f, 0.f, 0.f};
  v8f acc[2][4];
#pragma unroll
  for (int i = 0; i < 2; ++i)
#pragma unroll
    for (int j = 0; j < 4; ++j) acc[i][j] = vzero;

#if USE_TDM
  __shared__ u16 sA[2][TILE_ELEMS];
  __shared__ u16 sB[2][TILE_ELEMS];
  const unsigned ldsA = lds_off_of(&sA[0][0]);
  const unsigned ldsB = lds_off_of(&sB[0][0]);
  const u16* gA0 = A  + (size_t)bm * K;
  const u16* gB0 = Bw + (size_t)bn * K;
  const unsigned rowsB = (unsigned)((N - bn) < 128 ? (N - bn) : 128);
  const int nIter = K >> 5;

  if (wave == 0) {  // prologue: DMA slice 0 into buffer 0
    tdm_load_tile(ldsA, gA0, 128u, (unsigned)K, (unsigned)K);
    tdm_load_tile(ldsB, gB0, rowsB, (unsigned)K, (unsigned)K);
  }
  for (int it = 0; it < nIter; ++it) {
    if (wave == 0) {
      if (it + 1 < nIter) {  // DMA next slice into other buffer, keep 1 in flight
        const unsigned nb = (unsigned)((it + 1) & 1) * TILE_BYTES;
        const int kn = (it + 1) * 32;
        tdm_load_tile(ldsA + nb, gA0 + kn, 128u, (unsigned)(K - kn), (unsigned)K);
        tdm_load_tile(ldsB + nb, gB0 + kn, rowsB, (unsigned)(K - kn), (unsigned)K);
        __builtin_amdgcn_s_wait_tensorcnt(2);
      } else {
        __builtin_amdgcn_s_wait_tensorcnt(0);
      }
    }
    __syncthreads();
    const u16* sAb = &sA[it & 1][0];
    const u16* sBb = &sB[it & 1][0];

    FragAB a[2], b[4];
#pragma unroll
    for (int i = 0; i < 2; ++i) {
      const u16* pa = &sAb[(wm + i * 16 + l15) * 32 + lh * 8];
      a[i].f[0] = *(const v4f*)pa;
      a[i].f[1] = *(const v4f*)(pa + 16);
    }
#pragma unroll
    for (int j = 0; j < 4; ++j) {
      const u16* pb = &sBb[(wn + j * 16 + l15) * 32 + lh * 8];
      b[j].f[0] = *(const v4f*)pb;
      b[j].f[1] = *(const v4f*)(pb + 16);
    }
#pragma unroll
    for (int i = 0; i < 2; ++i)
#pragma unroll
      for (int j = 0; j < 4; ++j)
        acc[i][j] = __builtin_amdgcn_wmma_f32_16x16x32_bf16(
            false, a[i].v, false, b[j].v, (short)0, acc[i][j], false, false);
    __syncthreads();
  }
#else  // -------- fallback: synchronous cooperative staging --------
  __shared__ u16 sAs[TILE_ELEMS];
  __shared__ u16 sBs[TILE_ELEMS];
  const int ar = tid >> 1;
  const int ac = (tid & 1) * 16;
  for (int k0 = 0; k0 < K; k0 += 32) {
    const u16* ga = A + (size_t)(bm + ar) * K + k0 + ac;
    *(v4f*)&sAs[ar * 32 + ac]     = *(const v4f*)(ga);
    *(v4f*)&sAs[ar * 32 + ac + 8] = *(const v4f*)(ga + 8);
    if (bn + ar < N) {
      const u16* gb = Bw + (size_t)(bn + ar) * K + k0 + ac;
      *(v4f*)&sBs[ar * 32 + ac]     = *(const v4f*)(gb);
      *(v4f*)&sBs[ar * 32 + ac + 8] = *(const v4f*)(gb + 8);
    } else {
      const v4f z = {0.f, 0.f, 0.f, 0.f};
      *(v4f*)&sBs[ar * 32 + ac] = z;
      *(v4f*)&sBs[ar * 32 + ac + 8] = z;
    }
    __syncthreads();
    FragAB a[2], b[4];
#pragma unroll
    for (int i = 0; i < 2; ++i) {
      const u16* pa = &sAs[(wm + i * 16 + l15) * 32 + lh * 8];
      a[i].f[0] = *(const v4f*)pa;
      a[i].f[1] = *(const v4f*)(pa + 16);
    }
#pragma unroll
    for (int j = 0; j < 4; ++j) {
      const u16* pb = &sBs[(wn + j * 16 + l15) * 32 + lh * 8];
      b[j].f[0] = *(const v4f*)pb;
      b[j].f[1] = *(const v4f*)(pb + 16);
    }
#pragma unroll
    for (int i = 0; i < 2; ++i)
#pragma unroll
      for (int j = 0; j < 4; ++j)
        acc[i][j] = __builtin_amdgcn_wmma_f32_16x16x32_bf16(
            false, a[i].v, false, b[j].v, (short)0, acc[i][j], false, false);
    __syncthreads();
  }
#endif

  // Epilogue: D layout = VGPR r, lanes 0-15 -> M=r, lanes 16-31 -> M=r+8
#pragma unroll
  for (int i = 0; i < 2; ++i) {
    const int rb = bm + wm + i * 16 + lh * 8;
#pragma unroll
    for (int j = 0; j < 4; ++j) {
      const int col = bn + wn + j * 16 + l15;
      if (col >= N) continue;
      const float bval = bias ? bias[col] : 0.0f;
#pragma unroll
      for (int r = 0; r < 8; ++r) {
        const int row = rb + r;
        const float v = acc[i][j][r] + bval;
        if (EPI == 0) {
          ((float*)OutP)[(size_t)row * ldo + col] = v;
        } else if (EPI == 1) {
          ((u16*)OutP)[(size_t)row * ldo + col] = f2bf(v);
        } else if (EPI == 2) {
          ((float*)OutP)[(size_t)row * ldo + col] += v;
        } else {
          const float g = 0.5f * v * (1.0f + erff(v * 0.70710678f));
          ((u16*)OutP)[(size_t)row * ldo + col] = f2bf(g);
        }
      }
    }
  }
}

// ---------------------------------------------------------------------------
// Head permutes: [B*T, C] bf16 -> [B,H,T,D]   /  -> [B,H,D,T] (V transposed)
// ---------------------------------------------------------------------------
__global__ void perm_qk_k(const u16* __restrict__ in, u16* __restrict__ out) {
  for (int i = blockIdx.x * blockDim.x + threadIdx.x; i < Mm * Cc;
       i += gridDim.x * blockDim.x) {
    int c = i & (Cc - 1), bt = i / Cc;
    int b = bt / Tt, t = bt & (Tt - 1);
    int h = c / Dd, d = c & (Dd - 1);
    out[((size_t)(b * Hh + h) * Tt + t) * Dd + d] = in[i];
  }
}
__global__ void perm_vt_k(const u16* __restrict__ in, u16* __restrict__ out) {
  for (int i = blockIdx.x * blockDim.x + threadIdx.x; i < Mm * Cc;
       i += gridDim.x * blockDim.x) {
    int c = i & (Cc - 1), bt = i / Cc;
    int b = bt / Tt, t = bt & (Tt - 1);
    int h = c / Dd, d = c & (Dd - 1);
    out[((size_t)(b * Hh + h) * Dd + d) * Tt + t] = in[i];
  }
}

// ---------------------------------------------------------------------------
// Attention scores: S[bh,q,k] = (Q . K) * 1/sqrt(D), WMMA over D=64 (2 slices).
// Wave tile 16x16; tiles fully above the causal diagonal are skipped.
// ---------------------------------------------------------------------------
__global__ __launch_bounds__(256)
void attn_scores_k(const u16* __restrict__ Q, const u16* __restrict__ Kp,
                   float* __restrict__ S) {
  const int bh = blockIdx.z;
  const int q0 = blockIdx.y * 16;
  const int lane = threadIdx.x & 31, wave = threadIdx.x >> 5;
  const int k0 = blockIdx.x * 128 + wave * 16;
  if (k0 > q0 + 15) return;          // wave-uniform: EXEC stays all-ones
  const int l15 = lane & 15, lh = lane >> 4;
  const u16* qb = Q  + ((size_t)bh * Tt + q0 + l15) * Dd;
  const u16* kb = Kp + ((size_t)bh * Tt + k0 + l15) * Dd;
  v8f acc = {0.f, 0.f, 0.f, 0.f, 0.f, 0.f, 0.f, 0.f};
#pragma unroll
  for (int c = 0; c < 2; ++c) {      // D = 64 = 2 x K32
    FragAB a, b;
    const u16* pa = qb + c * 32 + lh * 8;
    a.f[0] = *(const v4f*)pa;
    a.f[1] = *(const v4f*)(pa + 16);
    const u16* pb = kb + c * 32 + lh * 8;
    b.f[0] = *(const v4f*)pb;
    b.f[1] = *(const v4f*)(pb + 16);
    acc = __builtin_amdgcn_wmma_f32_16x16x32_bf16(false, a.v, false, b.v,
                                                  (short)0, acc, false, false);
  }
  float* out = S + (size_t)bh * Tt * Tt;
#pragma unroll
  for (int r = 0; r < 8; ++r) {
    const int row = q0 + lh * 8 + r;
    out[(size_t)row * Tt + k0 + l15] = acc[r] * 0.125f;  // 1/sqrt(64)
  }
}

// ---------------------------------------------------------------------------
// Causal softmax: f32 scores row -> bf16 probs row (zeros past the diagonal)
// ---------------------------------------------------------------------------
__global__ __launch_bounds__(256)
void softmax_k(const float* __restrict__ S, u16* __restrict__ P) {
  const int row = blockIdx.x, bh = blockIdx.y;
  const int n = row + 1;
  const float* s = S + ((size_t)bh * Tt + row) * Tt;
  u16* p = P + ((size_t)bh * Tt + row) * Tt;
  __shared__ float red[256];
  const int tid = threadIdx.x;
  float mx = -3.4e38f;
  for (int i = tid; i < n; i += 256) mx = fmaxf(mx, s[i]);
  red[tid] = mx; __syncthreads();
  for (int st = 128; st > 0; st >>= 1) {
    if (tid < st) red[tid] = fmaxf(red[tid], red[tid + st]);
    __syncthreads();
  }
  mx = red[0]; __syncthreads();
  float sum = 0.f;
  for (int i = tid; i < n; i += 256) sum += __expf(s[i] - mx);
  red[tid] = sum; __syncthreads();
  for (int st = 128; st > 0; st >>= 1) {
    if (tid < st) red[tid] += red[tid + st];
    __syncthreads();
  }
  const float inv = 1.0f / red[0];
  for (int i = tid; i < Tt; i += 256)
    p[i] = (i < n) ? f2bf(__expf(s[i] - mx) * inv) : (u16)0;
}

// ---------------------------------------------------------------------------
// y = probs @ V : A = probs (bf16, masked zeros), B cols come from Vt[B,H,D,T]
// so B-fragments are contiguous. 2 waves/block, each wave 16 rows x 32 cols.
// ---------------------------------------------------------------------------
__global__ __launch_bounds__(64)
void attn_av_k(const u16* __restrict__ P, const u16* __restrict__ Vt,
               u16* __restrict__ Y) {
  const int bh = blockIdx.y, q0 = blockIdx.x * 16;
  const int lane = threadIdx.x & 31, wave = threadIdx.x >> 5;
  const int n0 = wave * 32;
  const int l15 = lane & 15, lh = lane >> 4;
  const u16* pb = P + ((size_t)bh * Tt + q0 + l15) * Tt;
  const v8f vzero = {0.f, 0.f, 0.f, 0.f, 0.f, 0.f, 0.f, 0.f};
  v8f acc[2] = {vzero, vzero};
  for (int k0 = 0; k0 < Tt; k0 += 32) {
    FragAB a;
    const u16* pa = pb + k0 + lh * 8;
    a.f[0] = *(const v4f*)pa;
    a.f[1] = *(const v4f*)(pa + 16);
#pragma unroll
    for (int j = 0; j < 2; ++j) {
      FragAB b;
      const u16* pv = Vt + ((size_t)bh * Dd + n0 + j * 16 + l15) * Tt + k0 + lh * 8;
      b.f[0] = *(const v4f*)pv;
      b.f[1] = *(const v4f*)(pv + 16);
      acc[j] = __builtin_amdgcn_wmma_f32_16x16x32_bf16(
          false, a.v, false, b.v, (short)0, acc[j], false, false);
    }
  }
  const int b_ = bh >> 4, h_ = bh & 15;
#pragma unroll
  for (int j = 0; j < 2; ++j)
#pragma unroll
    for (int r = 0; r < 8; ++r) {
      const int t = q0 + lh * 8 + r;
      const int col = h_ * Dd + n0 + j * 16 + l15;
      Y[(size_t)(b_ * Tt + t) * Cc + col] = f2bf(acc[j][r]);
    }
}

// ---------------------------------------------------------------------------
// Orchestration
// ---------------------------------------------------------------------------
extern "C" void kernel_launch(void* const* d_in, const int* in_sizes, int n_in,
                              void* d_out, int out_size, void* d_ws, size_t ws_size,
                              hipStream_t stream) {
  (void)in_sizes; (void)n_in; (void)out_size; (void)ws_size;
  const int*   idx   = (const int*)  d_in[0];
  const float* tok   = (const float*)d_in[1];
  const float* pos   = (const float*)d_in[2];
  const float* Wq    = (const float*)d_in[3];
  const float* bq    = (const float*)d_in[4];
  const float* Wk    = (const float*)d_in[5];
  const float* bk    = (const float*)d_in[6];
  const float* Wv    = (const float*)d_in[7];
  const float* bv    = (const float*)d_in[8];
  const float* Wp    = (const float*)d_in[9];
  const float* bp    = (const float*)d_in[10];
  const float* ln1w  = (const float*)d_in[11];
  const float* ln1b  = (const float*)d_in[12];
  const float* ln2w  = (const float*)d_in[13];
  const float* ln2b  = (const float*)d_in[14];
  const float* W1    = (const float*)d_in[15];
  const float* b1    = (const float*)d_in[16];
  const float* W2    = (const float*)d_in[17];
  const float* b2    = (const float*)d_in[18];
  const float* lnfw  = (const float*)d_in[19];
  const float* lnfb  = (const float*)d_in[20];
  const float* headw = (const float*)d_in[21];

  char* p = (char*)d_ws;
  auto take = [&](size_t bytes) -> void* {
    void* r = (void*)p;
    p += (bytes + 255) & ~(size_t)255;
    return r;
  };
  const size_t CCsQ = (size_t)Cc * Cc;

  u16* wq_bf   = (u16*)take(2 * (size_t)Ll * CCsQ);
  u16* wk_bf   = (u16*)take(2 * (size_t)Ll * CCsQ);
  u16* wv_bf   = (u16*)take(2 * (size_t)Ll * CCsQ);
  u16* wp_bf   = (u16*)take(2 * (size_t)Ll * CCsQ);
  u16* w1_bf   = (u16*)take(2 * (size_t)Ll * 4 * CCsQ);
  u16* w2_bf   = (u16*)take(2 * (size_t)Ll * 4 * CCsQ);
  u16* head_bf = (u16*)take(2 * (size_t)Vv * Cc);
  float* x     = (float*)take(4 * (size_t)Mm * Cc);
  u16* h       = (u16*)take(2 * (size_t)Mm * Cc);
  u16* qlin    = (u16*)take(2 * (size_t)Mm * Cc);
  u16* klin    = (u16*)take(2 * (size_t)Mm * Cc);
  u16* vlin    = (u16*)take(2 * (size_t)Mm * Cc);
  u16* qp      = (u16*)take(2 * (size_t)Mm * Cc);
  u16* kp      = (u16*)take(2 * (size_t)Mm * Cc);
  u16* vt      = (u16*)take(2 * (size_t)Mm * Cc);
  u16* y       = (u16*)take(2 * (size_t)Mm * Cc);
  u16* mact    = (u16*)take(2 * (size_t)Mm * 4 * Cc);
  float* scores= (float*)take(4 * (size_t)BH * Tt * Tt);
  u16* probs   = (u16*)take(2 * (size_t)BH * Tt * Tt);

  // One-time (per launch) weight conversions f32 -> bf16
  cvt_k<<<4096, 256, 0, stream>>>(Wq, wq_bf, (size_t)Ll * CCsQ);
  cvt_k<<<4096, 256, 0, stream>>>(Wk, wk_bf, (size_t)Ll * CCsQ);
  cvt_k<<<4096, 256, 0, stream>>>(Wv, wv_bf, (size_t)Ll * CCsQ);
  cvt_k<<<4096, 256, 0, stream>>>(Wp, wp_bf, (size_t)Ll * CCsQ);
  cvt_k<<<4096, 256, 0, stream>>>(W1, w1_bf, (size_t)Ll * 4 * CCsQ);
  cvt_k<<<4096, 256, 0, stream>>>(W2, w2_bf, (size_t)Ll * 4 * CCsQ);
  cvt_k<<<4096, 256, 0, stream>>>(headw, head_bf, (size_t)Vv * Cc);

  embed_k<<<2048, 256, 0, stream>>>(idx, tok, pos, x);

  const dim3 gN1(Cc / 128, Mm / 128);       // N=1024 GEMMs: (8,16)
  const dim3 gN4(4 * Cc / 128, Mm / 128);   // N=4096 GEMM : (32,16)

  for (int l = 0; l < Ll; ++l) {
    const u16* wqb = wq_bf + (size_t)l * CCsQ;
    const u16* wkb = wk_bf + (size_t)l * CCsQ;
    const u16* wvb = wv_bf + (size_t)l * CCsQ;
    const u16* wpb = wp_bf + (size_t)l * CCsQ;
    const u16* w1b = w1_bf + (size_t)l * 4 * CCsQ;
    const u16* w2b = w2_bf + (size_t)l * 4 * CCsQ;

    ln_k<<<Mm, 256, 0, stream>>>(x, ln1w + l * Cc, ln1b + l * Cc, h);

    gemm_bf16_k<1><<<gN1, 256, 0, stream>>>(h, wqb, bq + l * Cc, qlin, Mm, Cc, Cc, Cc);
    gemm_bf16_k<1><<<gN1, 256, 0, stream>>>(h, wkb, bk + l * Cc, klin, Mm, Cc, Cc, Cc);
    gemm_bf16_k<1><<<gN1, 256, 0, stream>>>(h, wvb, bv + l * Cc, vlin, Mm, Cc, Cc, Cc);

    perm_qk_k<<<2048, 256, 0, stream>>>(qlin, qp);
    perm_qk_k<<<2048, 256, 0, stream>>>(klin, kp);
    perm_vt_k<<<2048, 256, 0, stream>>>(vlin, vt);

    attn_scores_k<<<dim3(Tt / 128, Tt / 16, BH), 256, 0, stream>>>(qp, kp, scores);
    softmax_k<<<dim3(Tt, BH), 256, 0, stream>>>(scores, probs);
    attn_av_k<<<dim3(Tt / 16, BH), 64, 0, stream>>>(probs, vt, y);

    gemm_bf16_k<2><<<gN1, 256, 0, stream>>>(y, wpb, bp + l * Cc, x, Mm, Cc, Cc, Cc);

    ln_k<<<Mm, 256, 0, stream>>>(x, ln2w + l * Cc, ln2b + l * Cc, h);
    gemm_bf16_k<3><<<gN4, 256, 0, stream>>>(h, w1b, b1 + (size_t)l * 4 * Cc, mact,
                                            Mm, 4 * Cc, Cc, 4 * Cc);
    gemm_bf16_k<2><<<gN1, 256, 0, stream>>>(mact, w2b, b2 + l * Cc, x,
                                            Mm, Cc, 4 * Cc, Cc);
  }

  ln_k<<<Mm, 256, 0, stream>>>(x, lnfw, lnfb, h);
  gemm_bf16_k<0><<<dim3((Vv + 127) / 128, Mm / 128), 256, 0, stream>>>(
      h, head_bf, nullptr, d_out, Mm, Vv, Cc, Vv);
}
